// GCNLayer_6854767805296
// MI455X (gfx1250) — matrix operand order repaired and checked by reading, not compiled, not run on hardware
//
#include <hip/hip_runtime.h>
#include <hip/hip_bf16.h>

// ---- CDNA5 WMMA vector types ----
typedef __attribute__((ext_vector_type(16))) __bf16 v16bf;
typedef __attribute__((ext_vector_type(8)))  __bf16 bf16x8;
typedef __attribute__((ext_vector_type(8)))  float  v8f;
typedef __attribute__((ext_vector_type(4)))  float  f32x4;

static constexpr int   NN   = 8192;  // nodes
static constexpr int   CIN  = 512;   // in features
static constexpr int   COUT = 512;   // out features
static constexpr float EPS  = 1e-8f;

static constexpr int LDB = 40;       // padded LDS row stride (bf16 elems): 80B -> conflict-free

// ------------------------------------------------------------------
// Kernel 1: dis[i] = rsqrt(sum_j adj[i][j] + eps)
// ------------------------------------------------------------------
__global__ void k_rowsum(const float* __restrict__ adj, float* __restrict__ dis) {
    __shared__ float red[256];
    const int row = blockIdx.x;
    const f32x4* p = (const f32x4*)(adj + (size_t)row * NN);
    float s = 0.f;
    #pragma unroll 4
    for (int i = threadIdx.x; i < NN / 4; i += 256) {
        f32x4 v = p[i];
        s += v.x + v.y + v.z + v.w;
    }
    red[threadIdx.x] = s;
    __syncthreads();
    for (int off = 128; off > 0; off >>= 1) {
        if ((int)threadIdx.x < off) red[threadIdx.x] += red[threadIdx.x + off];
        __syncthreads();
    }
    if (threadIdx.x == 0) dis[row] = rsqrtf(red[0] + EPS);
}

// ------------------------------------------------------------------
// Kernel 2: xsT[c][j] = bf16( dis[j] * x[j][c] )   (transposed, bf16)
// ------------------------------------------------------------------
__global__ void k_scale_transpose(const float* __restrict__ x,
                                  const float* __restrict__ dis,
                                  __bf16* __restrict__ xsT) {
    __shared__ float tile[64][65];
    const int tx = threadIdx.x & 63;
    const int ty = threadIdx.x >> 6;
    const int J0 = blockIdx.x * 64;
    const int C0 = blockIdx.y * 64;
    #pragma unroll
    for (int jj = ty; jj < 64; jj += 4)
        tile[jj][tx] = x[(size_t)(J0 + jj) * CIN + C0 + tx];
    __syncthreads();
    const float d = dis[J0 + tx];
    #pragma unroll
    for (int cc = ty; cc < 64; cc += 4)
        xsT[(size_t)(C0 + cc) * NN + J0 + tx] = (__bf16)(tile[tx][cc] * d);
}

// ------------------------------------------------------------------
// Kernel 3: h = dis[i] * (adj_bf16 @ xsT^T)  -> bf16 h [NN][CIN]
// wave tile 16M x 128N; block = 8 waves stacked in M (128M x 128N)
// B tile (32K x 128N, 8KB) staged once per WG in LDS, double-buffered;
// A fragments pipelined in registers from the adj stream.
// ------------------------------------------------------------------
__global__ void k_gemm_agg(const float* __restrict__ adj,
                           const __bf16* __restrict__ xsT,
                           const float* __restrict__ dis,
                           __bf16* __restrict__ h) {
    __shared__ __bf16 Bs[2][128 * LDB];   // 2 x 10KB

    const int t    = threadIdx.x;
    const int w    = t >> 5;
    const int lane = t & 31;
    const int g    = lane >> 4;
    const int ln   = lane & 15;
    const int rowbase = blockIdx.x * 128 + w * 16;
    const int colbase = blockIdx.y * 128;

    // cooperative B staging: 512 chunks of 8 bf16; thread t owns chunks 2t, 2t+1
    const int sn  = t >> 1;                 // row n (0..127), two threads per row
    const int sk  = (t & 1) * 16;           // this thread covers K [sk, sk+16)
    const __bf16* gB = xsT + (size_t)(colbase + sn) * NN + sk;
    __bf16* lB = &Bs[0][0] + sn * LDB + sk;

    v8f acc[8] = {};
    const float* arow = adj + (size_t)(rowbase + ln) * NN;

    // ---- prologue: stage tile 0, preload A(0) ----
    bf16x8 s0 = *(const bf16x8*)(gB);
    bf16x8 s1 = *(const bf16x8*)(gB + 8);
    *(bf16x8*)(lB) = s0;
    *(bf16x8*)(lB + 8) = s1;
    f32x4 acur[4];
    {
        const f32x4* p0 = (const f32x4*)(arow + g * 8);
        const f32x4* p1 = (const f32x4*)(arow + 16 + g * 8);
        acur[0] = p0[0]; acur[1] = p0[1]; acur[2] = p1[0]; acur[3] = p1[1];
    }

    constexpr int NSTEP = NN / 32;          // 256
    int buf = 0;
    for (int step = 0; step < NSTEP; ++step) {
        const int k0 = step * 32;
        __syncthreads();                    // Bs[buf] ready; Bs[buf^1] free

        f32x4 anext[4];
        bf16x8 n0, n1;
        if (step + 1 < NSTEP) {
            // issue next-step global loads a full iteration ahead
            n0 = *(const bf16x8*)(gB + (k0 + 32));
            n1 = *(const bf16x8*)(gB + (k0 + 32) + 8);
            const f32x4* p0 = (const f32x4*)(arow + k0 + 32 + g * 8);
            const f32x4* p1 = (const f32x4*)(arow + k0 + 48 + g * 8);
            anext[0] = p0[0]; anext[1] = p0[1]; anext[2] = p1[0]; anext[3] = p1[1];
            __builtin_prefetch(arow + k0 + 1024, 0, 0);   // keep adj streaming
        }

        // A fragment: cvt fp32 -> bf16 (ISA 16-bit A layout: K runs g*8, 16+g*8)
        v16bf a;
        #pragma unroll
        for (int i = 0; i < 4; ++i) {
            a[i]      = (__bf16)acur[0][i];
            a[4 + i]  = (__bf16)acur[1][i];
            a[8 + i]  = (__bf16)acur[2][i];
            a[12 + i] = (__bf16)acur[3][i];
        }

        // 8 N-tiles from LDS, shared A fragment
        const __bf16* bbase = &Bs[buf][0] + ln * LDB + g * 8;
        #pragma unroll
        for (int nt = 0; nt < 8; ++nt) {
            const __bf16* bp = bbase + nt * 16 * LDB;
            bf16x8 b0 = *(const bf16x8*)bp;
            bf16x8 b1 = *(const bf16x8*)(bp + 16);
            v16bf b;
            #pragma unroll
            for (int i = 0; i < 8; ++i) { b[i] = b0[i]; b[8 + i] = b1[i]; }
            acc[nt] = __builtin_amdgcn_wmma_f32_16x16x32_bf16(
                false, a, false, b, (short)0, acc[nt], false, false);
        }

        if (step + 1 < NSTEP) {
            __bf16* ldst = &Bs[buf ^ 1][0] + sn * LDB + sk;
            *(bf16x8*)(ldst) = n0;
            *(bf16x8*)(ldst + 8) = n1;
            acur[0] = anext[0]; acur[1] = anext[1];
            acur[2] = anext[2]; acur[3] = anext[3];
        }
        buf ^= 1;
    }

    // scale rows by dis[i], write bf16 h
    float dscale[8];
    #pragma unroll
    for (int r = 0; r < 8; ++r) dscale[r] = dis[rowbase + r + 8 * g];
    #pragma unroll
    for (int nt = 0; nt < 8; ++nt) {
        const int col = colbase + nt * 16 + ln;
        #pragma unroll
        for (int r = 0; r < 8; ++r) {
            const int row = rowbase + r + 8 * g;
            h[(size_t)row * CIN + col] = (__bf16)(acc[nt][r] * dscale[r]);
        }
    }
}

// ------------------------------------------------------------------
// Kernel 4: out = h @ W^T + b   (h bf16, W fp32 [OUT][IN] row-major)
// wave tile 16M x 64N; block = 8 waves stacked in M (128M x 64N)
// W tile (32K x 64N) converted to bf16 once per WG into LDS, double-buffered.
// ------------------------------------------------------------------
__global__ void k_gemm_out(const __bf16* __restrict__ h,
                           const float* __restrict__ W,
                           const float* __restrict__ bias,
                           float* __restrict__ out) {
    __shared__ __bf16 Ws[2][64 * LDB];    // 2 x 5KB

    const int t    = threadIdx.x;
    const int w    = t >> 5;
    const int lane = t & 31;
    const int g    = lane >> 4;
    const int ln   = lane & 15;
    const int rowbase = blockIdx.x * 128 + w * 16;
    const int colbase = blockIdx.y * 64;

    // cooperative staging: 256 chunks of 8 floats; thread t owns chunk t
    const int sn = t >> 2;                 // row n (0..63)
    const int sk = (t & 3) * 8;            // K offset
    const float* gW = W + (size_t)(colbase + sn) * CIN + sk;
    __bf16* lW = &Ws[0][0] + sn * LDB + sk;

    v8f acc[4] = {};
    const __bf16* arow = h + (size_t)(rowbase + ln) * CIN;

    // prologue
    {
        const f32x4* p = (const f32x4*)gW;
        f32x4 w0 = p[0], w1 = p[1];
        bf16x8 r;
        #pragma unroll
        for (int i = 0; i < 4; ++i) { r[i] = (__bf16)w0[i]; r[4 + i] = (__bf16)w1[i]; }
        *(bf16x8*)lW = r;
    }
    bf16x8 a0 = *(const bf16x8*)(arow + g * 8);
    bf16x8 a1 = *(const bf16x8*)(arow + 16 + g * 8);

    constexpr int NSTEP = CIN / 32;        // 16
    int buf = 0;
    for (int step = 0; step < NSTEP; ++step) {
        const int k0 = step * 32;
        __syncthreads();

        bf16x8 na0, na1, nw;
        if (step + 1 < NSTEP) {
            const f32x4* p = (const f32x4*)(gW + k0 + 32);
            f32x4 w0 = p[0], w1 = p[1];
            #pragma unroll
            for (int i = 0; i < 4; ++i) { nw[i] = (__bf16)w0[i]; nw[4 + i] = (__bf16)w1[i]; }
            na0 = *(const bf16x8*)(arow + k0 + 32 + g * 8);
            na1 = *(const bf16x8*)(arow + k0 + 48 + g * 8);
        }

        v16bf a;
        #pragma unroll
        for (int i = 0; i < 8; ++i) { a[i] = a0[i]; a[8 + i] = a1[i]; }

        const __bf16* bbase = &Ws[buf][0] + ln * LDB + g * 8;
        #pragma unroll
        for (int nt = 0; nt < 4; ++nt) {
            const __bf16* bp = bbase + nt * 16 * LDB;
            bf16x8 b0 = *(const bf16x8*)bp;
            bf16x8 b1 = *(const bf16x8*)(bp + 16);
            v16bf b;
            #pragma unroll
            for (int i = 0; i < 8; ++i) { b[i] = b0[i]; b[8 + i] = b1[i]; }
            acc[nt] = __builtin_amdgcn_wmma_f32_16x16x32_bf16(
                false, a, false, b, (short)0, acc[nt], false, false);
        }

        if (step + 1 < NSTEP) {
            *(bf16x8*)(&Ws[buf ^ 1][0] + sn * LDB + sk) = nw;
            a0 = na0; a1 = na1;
        }
        buf ^= 1;
    }

    #pragma unroll
    for (int nt = 0; nt < 4; ++nt) {
        const int col = colbase + nt * 16 + ln;
        const float bb = bias[col];
        #pragma unroll
        for (int r = 0; r < 8; ++r) {
            const int row = rowbase + r + 8 * g;
            out[(size_t)row * COUT + col] = acc[nt][r] + bb;
        }
    }
}

// ------------------------------------------------------------------
extern "C" void kernel_launch(void* const* d_in, const int* in_sizes, int n_in,
                              void* d_out, int out_size, void* d_ws, size_t ws_size,
                              hipStream_t stream) {
    (void)in_sizes; (void)n_in; (void)out_size; (void)ws_size;
    const float* x   = (const float*)d_in[0];   // [NN, CIN]
    const float* adj = (const float*)d_in[1];   // [NN, NN]
    const float* W   = (const float*)d_in[2];   // [COUT, CIN]
    const float* b   = (const float*)d_in[3];   // [COUT]
    float* out = (float*)d_out;                 // [NN, COUT]

    // workspace layout: dis (32KB f32) | xsT (8MB bf16) | h (8MB bf16)
    float*  dis = (float*)d_ws;
    __bf16* xsT = (__bf16*)((char*)d_ws + 32768);
    __bf16* h   = xsT + (size_t)CIN * NN;

    k_rowsum<<<NN, 256, 0, stream>>>(adj, dis);
    k_scale_transpose<<<dim3(NN / 64, CIN / 64), 256, 0, stream>>>(x, dis, xsT);
    k_gemm_agg<<<dim3(NN / 128, CIN / 128), 256, 0, stream>>>(adj, xsT, dis, h);
    k_gemm_out<<<dim3(NN / 128, COUT / 64), 256, 0, stream>>>(h, W, b, out);
}
